// EEGGAT_superpool_62663572849446
// MI455X (gfx1250) — matrix operand order repaired and checked by reading, not compiled, not run on hardware
//
#include <hip/hip_runtime.h>
#include <hip/hip_bf16.h>

// ---------------------------------------------------------------------------
// EEG-GAT superpool: 2x GATConv(4 heads, 64 ch) + supernode readout.
// GEMMs run on v_wmma_f32_16x16x32_bf16 with bf16x3 error compensation.
// Edge softmax/aggregation via fp32 atomics (L2-resident, scatter bound).
// ---------------------------------------------------------------------------

typedef __bf16 bf16_t;
typedef __attribute__((ext_vector_type(16))) __bf16 v16bf;
typedef __attribute__((ext_vector_type(8)))  float  v8f;
typedef __attribute__((ext_vector_type(4), may_alias)) unsigned int u32x4a;

struct Raw32 { u32x4a lo; u32x4a hi; };
static_assert(sizeof(Raw32) == 32, "Raw32 must match v16bf size");

// Problem constants (match reference file)
constexpr long NN    = 81920;           // nodes
constexpr long EE    = 1310720;         // edges (without self loops)
constexpr long ETOT  = EE + NN;         // + self loops
constexpr long GG    = 4096;            // graphs
constexpr int  NPER  = 20;
constexpr int  INDIM = 128;
constexpr int  D1    = 256;             // H1*HID
constexpr int  D2    = 256;             // H2*HID
constexpr int  HIDC  = 64;
constexpr int  NH    = 4;

// ---------------------------------------------------------------------------
// WMMA tile loaders.
// 16-bit A 16x32: lane g*16+m, elems 0..15 = K[g*16..g*16+15] of row m.
// 16-bit B 32x16: lane g*16+n, elems 0..15 = K[g*16..g*16+15] of column n.
// ---------------------------------------------------------------------------
__device__ __forceinline__ v16bf load_a_tile(const bf16_t* row, int k0, int g) {
  Raw32 r;
  r.lo = *(const u32x4a*)(row + k0 + g * 16);
  r.hi = *(const u32x4a*)(row + k0 + g * 16 + 8);
  return __builtin_bit_cast(v16bf, r);
}
__device__ __forceinline__ v16bf load_b_tile(const bf16_t* row, int k0, int g) {
  Raw32 r;
  r.lo = *(const u32x4a*)(row + k0 + g * 16);
  r.hi = *(const u32x4a*)(row + k0 + g * 16 + 8);
  return __builtin_bit_cast(v16bf, r);
}

// C[M,Ncols] = A[M,K] @ W[Ncols,K]^T, bf16x3 compensated, f32 accumulate.
__global__ void __launch_bounds__(256)
gemm_bf16x3(const bf16_t* __restrict__ Ahi, const bf16_t* __restrict__ Alo,
            const bf16_t* __restrict__ Whi, const bf16_t* __restrict__ Wlo,
            float* __restrict__ C, int M, int Ncols, int K) {
  const int wave = threadIdx.x >> 5;
  const int lane = threadIdx.x & 31;
  const int tile = blockIdx.x * 8 + wave;
  const int ntn  = Ncols >> 4;
  const int mt   = tile / ntn;
  const int nt   = tile % ntn;
  if (mt * 16 >= M) return;                 // uniform per wave: EXEC stays all-1
  const int g  = lane >> 4;
  const int ln = lane & 15;
  const bf16_t* arh = Ahi + (size_t)(mt * 16 + ln) * K;
  const bf16_t* arl = Alo + (size_t)(mt * 16 + ln) * K;
  const bf16_t* brh = Whi + (size_t)(nt * 16 + ln) * K;
  const bf16_t* brl = Wlo + (size_t)(nt * 16 + ln) * K;
  v8f acc = {0.f, 0.f, 0.f, 0.f, 0.f, 0.f, 0.f, 0.f};
  for (int k0 = 0; k0 < K; k0 += 32) {
    v16bf ah = load_a_tile(arh, k0, g);
    v16bf al = load_a_tile(arl, k0, g);
    v16bf wh = load_b_tile(brh, k0, g);
    v16bf wl = load_b_tile(brl, k0, g);
    acc = __builtin_amdgcn_wmma_f32_16x16x32_bf16(false, ah, false, wh, (short)0, acc, false, false);
    acc = __builtin_amdgcn_wmma_f32_16x16x32_bf16(false, ah, false, wl, (short)0, acc, false, false);
    acc = __builtin_amdgcn_wmma_f32_16x16x32_bf16(false, al, false, wh, (short)0, acc, false, false);
  }
  // C/D layout: VGPR r -> row (mt*16 + g*8 + r), col (nt*16 + ln)
  float* cp = C + (size_t)(mt * 16 + g * 8) * Ncols + (size_t)nt * 16 + ln;
#pragma unroll
  for (int r = 0; r < 8; ++r) cp[(size_t)r * Ncols] = acc[r];
}

// ---------------------------------------------------------------------------
// Helpers / elementwise kernels
// ---------------------------------------------------------------------------
__global__ void split_bf16(const float* __restrict__ in, bf16_t* __restrict__ hi,
                           bf16_t* __restrict__ lo, long n) {
  long i = (long)blockIdx.x * blockDim.x + threadIdx.x;
  if (i >= n) return;
  float f  = in[i];
  __bf16 h = (__bf16)f;
  hi[i] = h;
  lo[i] = (__bf16)(f - (float)h);
}

__global__ void fill_f32(float* __restrict__ p, float v, long n) {
  long i = (long)blockIdx.x * blockDim.x + threadIdx.x;
  if (i < n) p[i] = v;
}

// s_src[n,h] = sum_c h[n, h*64+c] * a_src[h,c]  (same for dst)
__global__ void scores_kernel(const float* __restrict__ hmat,
                              const float* __restrict__ a_src,
                              const float* __restrict__ a_dst,
                              float* __restrict__ ssrc, float* __restrict__ sdst) {
  long i = (long)blockIdx.x * blockDim.x + threadIdx.x;
  if (i >= NN * NH) return;
  long n = i >> 2;
  int  hh = (int)(i & 3);
  const float* hp = hmat + n * D1 + hh * HIDC;
  const float* as = a_src + hh * HIDC;
  const float* ad = a_dst + hh * HIDC;
  float s1 = 0.f, s2 = 0.f;
#pragma unroll 8
  for (int c = 0; c < HIDC; ++c) {
    float v = hp[c];
    s1 += v * as[c];
    s2 += v * ad[c];
  }
  ssrc[i] = s1;
  sdst[i] = s2;
}

__device__ __forceinline__ void edge_sd(const int* __restrict__ ei, long e,
                                        int& s, int& d) {
  if (e < EE) { s = ei[e]; d = ei[EE + e]; }
  else        { s = d = (int)(e - EE); }      // self loops appended
}

__device__ __forceinline__ float lrelu02(float x) { return x > 0.f ? x : 0.2f * x; }

__device__ __forceinline__ void atomicMaxF32(float* addr, float val) {
  if (val >= 0.f) atomicMax((int*)addr, __float_as_int(val));
  else            atomicMin((unsigned int*)addr, __float_as_uint(val));
}

__global__ void edge_max_kernel(const int* __restrict__ ei,
                                const float* __restrict__ ssrc,
                                const float* __restrict__ sdst,
                                float* __restrict__ emax) {
  long i = (long)blockIdx.x * blockDim.x + threadIdx.x;
  if (i >= ETOT * NH) return;
  long e = i >> 2;
  int hh = (int)(i & 3);
  int s, d;
  edge_sd(ei, e, s, d);
  float v = lrelu02(ssrc[(long)s * NH + hh] + sdst[(long)d * NH + hh]);
  atomicMaxF32(&emax[(long)d * NH + hh], v);
}

__global__ void edge_denom_kernel(const int* __restrict__ ei,
                                  const float* __restrict__ ssrc,
                                  const float* __restrict__ sdst,
                                  const float* __restrict__ emax,
                                  float* __restrict__ denom) {
  long i = (long)blockIdx.x * blockDim.x + threadIdx.x;
  if (i >= ETOT * NH) return;
  long e = i >> 2;
  int hh = (int)(i & 3);
  int s, d;
  edge_sd(ei, e, s, d);
  float v  = lrelu02(ssrc[(long)s * NH + hh] + sdst[(long)d * NH + hh]);
  float ee = expf(v - emax[(long)d * NH + hh]);
  atomicAdd(&denom[(long)d * NH + hh], ee);
}

// One 256-thread block per edge: agg[dst, c] += alpha[head(c)] * h[src, c]
__global__ void __launch_bounds__(256)
edge_aggregate_kernel(const int* __restrict__ ei,
                      const float* __restrict__ ssrc,
                      const float* __restrict__ sdst,
                      const float* __restrict__ emax,
                      const float* __restrict__ denom,
                      const float* __restrict__ hmat,
                      float* __restrict__ agg) {
  long e = blockIdx.x;
  int s, d;
  edge_sd(ei, e, s, d);
  __shared__ float alpha[NH];
  int t = threadIdx.x;
  if (t < NH) {
    float v  = lrelu02(ssrc[(long)s * NH + t] + sdst[(long)d * NH + t]);
    float ee = expf(v - emax[(long)d * NH + t]);
    alpha[t] = ee / (denom[(long)d * NH + t] + 1e-16f);
  }
  __syncthreads();
  float a = alpha[t >> 6];
  atomicAdd(&agg[(long)d * D1 + t], a * hmat[(long)s * D1 + t]);
}

// h1e = elu(agg + b); emit bf16 hi/lo planes for layer-2 GEMM
__global__ void bias_elu_split(const float* __restrict__ agg,
                               const float* __restrict__ bias,
                               bf16_t* __restrict__ hi, bf16_t* __restrict__ lo) {
  long i = (long)blockIdx.x * blockDim.x + threadIdx.x;
  if (i >= NN * (long)D1) return;
  int c = (int)(i & (D1 - 1));
  float v = agg[i] + bias[c];
  v = v > 0.f ? v : expm1f(v);
  __bf16 h = (__bf16)v;
  hi[i] = h;
  lo[i] = (__bf16)(v - (float)h);
}

// Readout: xs = elu(agg2[super]+b2); y = relu(xs@Wh^T+bh); out = y@Wc^T+bc
__global__ void __launch_bounds__(64)
head_kernel(const float* __restrict__ agg2, const float* __restrict__ b2,
            const float* __restrict__ Wh, const float* __restrict__ bh,
            const float* __restrict__ Wc, const float* __restrict__ bc,
            float* __restrict__ out) {
  int g = blockIdx.x;
  long node = (long)g * NPER + (NPER - 1);
  __shared__ float xs[D2];
  __shared__ float yv[HIDC];
  int t = threadIdx.x;  // 64 threads
  for (int i = t; i < D2; i += HIDC) {
    float v = agg2[node * D2 + i] + b2[i];
    xs[i] = v > 0.f ? v : expm1f(v);
  }
  __syncthreads();
  float acc = bh[t];
  for (int i = 0; i < D2; ++i) acc += xs[i] * Wh[t * D2 + i];
  acc = fmaxf(acc, 0.f);
  yv[t] = acc * Wc[t];
  __syncthreads();
  for (int s2 = 32; s2 > 0; s2 >>= 1) {
    if (t < s2) yv[t] += yv[t + s2];
    __syncthreads();
  }
  if (t == 0) out[g] = yv[0] + bc[0];
}

// ---------------------------------------------------------------------------
extern "C" void kernel_launch(void* const* d_in, const int* in_sizes, int n_in,
                              void* d_out, int out_size, void* d_ws, size_t ws_size,
                              hipStream_t stream) {
  (void)in_sizes; (void)n_in; (void)out_size; (void)ws_size;
  const float* x     = (const float*)d_in[0];
  const int*   ei    = (const int*)d_in[1];   // [2,E]
  const float* W1    = (const float*)d_in[3];
  const float* a1s   = (const float*)d_in[4];
  const float* a1d   = (const float*)d_in[5];
  const float* b1    = (const float*)d_in[6];
  const float* W2    = (const float*)d_in[7];
  const float* a2s   = (const float*)d_in[8];
  const float* a2d   = (const float*)d_in[9];
  const float* b2    = (const float*)d_in[10];
  const float* Wh    = (const float*)d_in[11];
  const float* bh    = (const float*)d_in[12];
  const float* Wc    = (const float*)d_in[13];
  const float* bc    = (const float*)d_in[14];
  float* out = (float*)d_out;

  // ---- carve workspace (256B aligned slices) ----
  char* p = (char*)d_ws;
  auto carve = [&](size_t bytes) -> char* {
    char* r = p;
    p += (bytes + 255) & ~(size_t)255;
    return r;
  };
  bf16_t* xhi  = (bf16_t*)carve(NN * INDIM * 2);
  bf16_t* xlo  = (bf16_t*)carve(NN * INDIM * 2);
  bf16_t* w1hi = (bf16_t*)carve((size_t)D1 * INDIM * 2);
  bf16_t* w1lo = (bf16_t*)carve((size_t)D1 * INDIM * 2);
  bf16_t* w2hi = (bf16_t*)carve((size_t)D2 * D1 * 2);
  bf16_t* w2lo = (bf16_t*)carve((size_t)D2 * D1 * 2);
  float*  hbuf = (float*)carve(NN * (size_t)D1 * 4);   // h1, later h2 (reused)
  float*  agg  = (float*)carve(NN * (size_t)D1 * 4);   // layer aggregate (reused)
  bf16_t* h1ehi = (bf16_t*)carve(NN * (size_t)D1 * 2);
  bf16_t* h1elo = (bf16_t*)carve(NN * (size_t)D1 * 2);
  float*  ssrc = (float*)carve(NN * NH * 4);
  float*  sdst = (float*)carve(NN * NH * 4);
  float*  emax = (float*)carve(NN * NH * 4);
  float*  den  = (float*)carve(NN * NH * 4);

  const int TB = 256;
  const long nElemEdgeH = ETOT * NH;                 // 5,570,560
  dim3 blkEdgeH((unsigned)((nElemEdgeH + TB - 1) / TB));
  dim3 blkNH((unsigned)((NN * NH + TB - 1) / TB));   // 1280
  dim3 blkND1((unsigned)((NN * (long)D1 + TB - 1) / TB)); // 81920
  const unsigned gemmBlocks = (unsigned)((NN / 16) * (D1 / 16) / 8); // 10240

  // ---- precision-split conversions ----
  split_bf16<<<(unsigned)((NN * INDIM + TB - 1) / TB), TB, 0, stream>>>(x, xhi, xlo, NN * INDIM);
  split_bf16<<<(unsigned)(((long)D1 * INDIM + TB - 1) / TB), TB, 0, stream>>>(W1, w1hi, w1lo, (long)D1 * INDIM);
  split_bf16<<<(unsigned)(((long)D2 * D1 + TB - 1) / TB), TB, 0, stream>>>(W2, w2hi, w2lo, (long)D2 * D1);

  // ================= Layer 1 =================
  gemm_bf16x3<<<gemmBlocks, TB, 0, stream>>>(xhi, xlo, w1hi, w1lo, hbuf, (int)NN, D1, INDIM);
  scores_kernel<<<blkNH, TB, 0, stream>>>(hbuf, a1s, a1d, ssrc, sdst);
  fill_f32<<<blkNH, TB, 0, stream>>>(emax, -__builtin_huge_valf(), NN * NH);
  fill_f32<<<blkNH, TB, 0, stream>>>(den, 0.f, NN * NH);
  fill_f32<<<blkND1, TB, 0, stream>>>(agg, 0.f, NN * (long)D1);
  edge_max_kernel<<<blkEdgeH, TB, 0, stream>>>(ei, ssrc, sdst, emax);
  edge_denom_kernel<<<blkEdgeH, TB, 0, stream>>>(ei, ssrc, sdst, emax, den);
  edge_aggregate_kernel<<<(unsigned)ETOT, TB, 0, stream>>>(ei, ssrc, sdst, emax, den, hbuf, agg);
  bias_elu_split<<<blkND1, TB, 0, stream>>>(agg, b1, h1ehi, h1elo);

  // ================= Layer 2 =================
  gemm_bf16x3<<<gemmBlocks, TB, 0, stream>>>(h1ehi, h1elo, w2hi, w2lo, hbuf, (int)NN, D2, D1);
  scores_kernel<<<blkNH, TB, 0, stream>>>(hbuf, a2s, a2d, ssrc, sdst);
  fill_f32<<<blkNH, TB, 0, stream>>>(emax, -__builtin_huge_valf(), NN * NH);
  fill_f32<<<blkNH, TB, 0, stream>>>(den, 0.f, NN * NH);
  fill_f32<<<blkND1, TB, 0, stream>>>(agg, 0.f, NN * (long)D1);
  edge_max_kernel<<<blkEdgeH, TB, 0, stream>>>(ei, ssrc, sdst, emax);
  edge_denom_kernel<<<blkEdgeH, TB, 0, stream>>>(ei, ssrc, sdst, emax, den);
  edge_aggregate_kernel<<<(unsigned)ETOT, TB, 0, stream>>>(ei, ssrc, sdst, emax, den, hbuf, agg);

  // ================= Readout =================
  head_kernel<<<(unsigned)GG, HIDC, 0, stream>>>(agg, b2, Wh, bh, Wc, bc, out);
}
